// AttentionCl_67353677136276
// MI455X (gfx1250) — compile-verified, matrix operand order checked by reading, not run on hardware
//
#include <hip/hip_runtime.h>
#include <hip/hip_bf16.h>

#define DIM_HEAD 32
#define NTOK     64
#define HEADS    16
#define DIM      512
#define QKVN     1536

#define BM 128
#define BN 128
#define BK 32
#define LDA 40   // padded LDS row stride (bf16 elems): 80B rows, conflict-free 16B frags
#define PLD 72   // padded LDS stride for attention P / V^T tiles

typedef __attribute__((ext_vector_type(16))) __bf16 bf16x16;
typedef __attribute__((ext_vector_type(8)))  __bf16 bf16x8;
typedef __attribute__((ext_vector_type(4)))  __bf16 bf16x4;
typedef __attribute__((ext_vector_type(8)))  float  f32x8;
typedef __attribute__((ext_vector_type(4)))  float  f32x4;
typedef __attribute__((ext_vector_type(4)))  unsigned int u32x4;
typedef __attribute__((ext_vector_type(8)))  int    i32x8;
typedef __attribute__((ext_vector_type(4)))  int    i32x4;

#ifndef __has_builtin
#define __has_builtin(x) 0
#endif
#if __has_builtin(__builtin_amdgcn_tensor_load_to_lds)
#define USE_TDM 1
#else
#define USE_TDM 0
#endif

__device__ __forceinline__ bf16x16 frag_cat(bf16x8 lo, bf16x8 hi) {
    return __builtin_shufflevector(lo, hi, 0,1,2,3,4,5,6,7,8,9,10,11,12,13,14,15);
}

__device__ __forceinline__ f32x8 wmma_bf16(bf16x16 a, bf16x16 b, f32x8 c) {
    // D = A(16x32) * B(32x16) + C, fp32 accumulate
    return __builtin_amdgcn_wmma_f32_16x16x32_bf16(false, a, false, b, (short)0, c,
                                                   false, false);
}

#if USE_TDM
// Generic 2-D TDM load: tile (tile_d0 elems x tile_d1 rows) from a tensor with
// row stride stride0 (all in data_size units) into LDS, optional LDS padding.
__device__ __forceinline__ void tdm_load_2d(unsigned lds_addr, const void* gaddr,
                                            unsigned ds_code,          // 1=2B, 2=4B
                                            unsigned tile_d0, unsigned tile_d1,
                                            unsigned long long tensor_d0,
                                            unsigned long long tensor_d1,
                                            unsigned long long stride0,
                                            unsigned pad_en, unsigned pad_interval,
                                            unsigned pad_amount) {
    unsigned long long ga = (unsigned long long)gaddr;
    u32x4 g0;
    g0[0] = 1u;                                           // count=1, user mode
    g0[1] = lds_addr;
    g0[2] = (unsigned)ga;
    g0[3] = (unsigned)((ga >> 32) & 0x01FFFFFFull) | 0x80000000u;  // addr hi | type=2
    i32x8 g1;
    g1[0] = (int)((ds_code << 16) | (pad_en << 20) | (pad_interval << 22) |
                  (pad_amount << 25));
    g1[1] = (int)((tensor_d0 & 0xFFFFull) << 16);
    g1[2] = (int)(((tensor_d0 >> 16) & 0xFFFFull) | ((tensor_d1 & 0xFFFFull) << 16));
    g1[3] = (int)(((tensor_d1 >> 16) & 0xFFFFull) | ((unsigned long long)tile_d0 << 16));
    g1[4] = (int)tile_d1;                                 // tile_dim1 | tile_dim2=0
    g1[5] = (int)(stride0 & 0xFFFFFFFFull);               // tensor_dim0_stride lo32
    g1[6] = (int)((stride0 >> 32) & 0xFFFFull);           // stride hi16
    g1[7] = 0;
    i32x4 g2 = {1, 1, 0, 0};                              // tensor_dim2=1, dim3=1
    i32x4 g3 = {0, 0, 0, 0};
#if defined(__clang_major__) && __clang_major__ >= 23
    i32x8 g4 = {0, 0, 0, 0, 0, 0, 0, 0};
    __builtin_amdgcn_tensor_load_to_lds(g0, g1, g2, g3, g4, 0);
#else
    __builtin_amdgcn_tensor_load_to_lds(g0, g1, g2, g3, 0);
#endif
}

__device__ __forceinline__ void tdm_wait0() {
#if __has_builtin(__builtin_amdgcn_s_wait_tensorcnt)
    __builtin_amdgcn_s_wait_tensorcnt(0);
#else
    asm volatile("s_wait_tensorcnt 0x0" ::: "memory");
#endif
    asm volatile("" ::: "memory");   // compiler barrier: TDM wrote LDS
}
#endif // USE_TDM

// ---------------------------------------------------------------------------
// Prep: weights -> bf16, dense rel-pos bias table [16][64][64]
// ---------------------------------------------------------------------------
__global__ void prep_kernel(const float* __restrict__ wqkv,
                            const float* __restrict__ wproj,
                            const float* __restrict__ table,
                            __bf16* __restrict__ wqkv_bf,
                            __bf16* __restrict__ wproj_bf,
                            float* __restrict__ bias) {
    int gid = blockIdx.x * blockDim.x + threadIdx.x;
    const int NQ = DIM * QKVN;
    const int NP = DIM * DIM;
    const int NB = HEADS * NTOK * NTOK;
    if (gid < NQ) {
        wqkv_bf[gid] = (__bf16)wqkv[gid];
    } else if (gid < NQ + NP) {
        int i = gid - NQ;
        wproj_bf[i] = (__bf16)wproj[i];
    } else if (gid < NQ + NP + NB) {
        int i = gid - NQ - NP;
        int h = i >> 12;
        int n = (i >> 6) & 63;
        int m = i & 63;
        int rn = n >> 3, cn = n & 7, rm = m >> 3, cm = m & 7;
        bias[i] = table[h * 225 + (rn - rm + 7) * 15 + (cn - cm + 7)];
    }
}

// ---------------------------------------------------------------------------
// QKV GEMM: (Bwin*64, 512) x (512, 1536) in bf16 WMMA, scatter into q/k/v
// layout [bwin][head][tok][d] (bf16). Scale folded into q, bias added.
// Grid: x = N-chunks (fast, shares A tile via L2), y = M-tiles.
// ---------------------------------------------------------------------------
__global__ __launch_bounds__(256)
void qkv_gemm_kernel(const float* __restrict__ x,
                     const __bf16* __restrict__ wqkv_bf,
                     const float* __restrict__ bqkv,
                     __bf16* __restrict__ qws,
                     __bf16* __restrict__ kws,
                     __bf16* __restrict__ vws) {
    __shared__ __bf16 As[BM * LDA];
    __shared__ __bf16 Bs[BN * LDA];

    const int tid  = threadIdx.x;
    const int lane = tid & 31;
    const int wid  = tid >> 5;
    const int m0   = blockIdx.y * BM;
    const int n0   = blockIdx.x * BN;
    const int wm   = (wid & 3) * 32;   // wave row offset in tile
    const int wn   = (wid >> 2) * 64;  // wave col offset in tile
    const int r    = lane & 15;
    const int kg   = (lane >> 4) * 8;
    const int hih  = lane >> 4;

    f32x8 acc[2][4] = {};

    for (int k0 = 0; k0 < DIM; k0 += BK) {
        // A tile: 128x32 fp32 -> bf16, 16B global loads; prefetch next slab
        #pragma unroll
        for (int it = 0; it < 4; ++it) {
            int c   = tid + it * 256;      // 0..1023
            int row = c >> 3;
            int kc  = (c & 7) * 4;
            const float* src = x + (size_t)(m0 + row) * DIM + k0 + kc;
            f32x4 v = *(const f32x4*)src;
            __builtin_prefetch(src + BK, 0, 3);   // speculative: OOB is dropped
            bf16x4 bv;
            bv[0] = (__bf16)v[0]; bv[1] = (__bf16)v[1];
            bv[2] = (__bf16)v[2]; bv[3] = (__bf16)v[3];
            *(bf16x4*)&As[row * LDA + kc] = bv;
        }
        // B tile transposed into LDS: Bs[n][k] = W[k0+k][n0+n]
        #pragma unroll
        for (int it = 0; it < 4; ++it) {
            int c  = tid + it * 256;
            int k  = c >> 5;               // 0..31
            int nb = (c & 31) * 4;
            bf16x4 bv = *(const bf16x4*)(wqkv_bf + (size_t)(k0 + k) * QKVN + n0 + nb);
            Bs[(nb + 0) * LDA + k] = bv[0];
            Bs[(nb + 1) * LDA + k] = bv[1];
            Bs[(nb + 2) * LDA + k] = bv[2];
            Bs[(nb + 3) * LDA + k] = bv[3];
        }
        __syncthreads();

        bf16x16 af[2], bfr[4];
        #pragma unroll
        for (int i = 0; i < 2; ++i) {
            int row = wm + i * 16 + r;
            bf16x8 lo = *(const bf16x8*)&As[row * LDA + kg];
            bf16x8 hi = *(const bf16x8*)&As[row * LDA + 16 + kg];
            af[i] = frag_cat(lo, hi);
        }
        #pragma unroll
        for (int j = 0; j < 4; ++j) {
            int col = wn + j * 16 + r;
            bf16x8 lo = *(const bf16x8*)&Bs[col * LDA + kg];
            bf16x8 hi = *(const bf16x8*)&Bs[col * LDA + 16 + kg];
            bfr[j] = frag_cat(lo, hi);
        }
        #pragma unroll
        for (int i = 0; i < 2; ++i)
            #pragma unroll
            for (int j = 0; j < 4; ++j)
                acc[i][j] = wmma_bf16(af[i], bfr[j], acc[i][j]);
        __syncthreads();
    }

    const float qscale = 0.17677669529663687f;   // 1/sqrt(32)
    #pragma unroll
    for (int i = 0; i < 2; ++i) {
        #pragma unroll
        for (int j = 0; j < 4; ++j) {
            int col   = n0 + wn + j * 16 + r;
            int sel   = col >> 9;
            int rem   = col & 511;
            int h     = rem >> 5;
            int d     = rem & 31;
            __bf16* dst = (sel == 0) ? qws : ((sel == 1) ? kws : vws);
            float bv  = bqkv[col];
            float sc  = (sel == 0) ? qscale : 1.0f;
            #pragma unroll
            for (int v = 0; v < 8; ++v) {
                int mrow = m0 + wm + i * 16 + 8 * hih + v;
                int bwin = mrow >> 6;
                int tok  = mrow & 63;
                float val = (acc[i][j][v] + bv) * sc;
                dst[((size_t)(bwin * HEADS + h) * NTOK + tok) * DIM_HEAD + d] = (__bf16)val;
            }
        }
    }
}

// ---------------------------------------------------------------------------
// Attention: one wave per (window, head). S = q k^T (+bias), softmax, P v.
// Bias tile DMA'd to LDS by the TDM, overlapped with the S WMMAs.
// ---------------------------------------------------------------------------
__global__ __launch_bounds__(32)
void attn_kernel(const __bf16* __restrict__ qws,
                 const __bf16* __restrict__ kws,
                 const __bf16* __restrict__ vws,
                 const float* __restrict__ bias,
                 __bf16* __restrict__ aout) {
    __shared__ __bf16 Ps[NTOK * PLD];       // softmax probs, row-major
    __shared__ __bf16 Vt[DIM_HEAD * PLD];   // V transposed: Vt[d][tok]
#if USE_TDM
    __shared__ float  Bls[NTOK * NTOK];     // bias tile (16KB)
#endif

    const int lane = threadIdx.x;
    const int bh   = blockIdx.x;
    const int b    = bh >> 4;
    const int h    = bh & 15;
    const size_t base = (size_t)bh * NTOK * DIM_HEAD;
    const int r   = lane & 15;
    const int kg  = (lane >> 4) * 8;
    const int hih = lane >> 4;

    const float* bp = bias + (size_t)h * NTOK * NTOK;
#if USE_TDM
    // Kick off the bias DMA first; it completes under the q.k^T WMMAs.
    tdm_load_2d((unsigned)(unsigned long long)&Bls[0], bp,
                /*ds=*/2, /*tile_d0=*/NTOK, /*tile_d1=*/NTOK,
                /*tensor_d0=*/NTOK, /*tensor_d1=*/NTOK, /*stride0=*/NTOK,
                0, 0, 0);
#endif

    // Stage V^T into LDS
    for (int t = lane; t < NTOK; t += 32) {
        const __bf16* src = vws + base + (size_t)t * DIM_HEAD;
        #pragma unroll
        for (int c8 = 0; c8 < 4; ++c8) {
            bf16x8 vv = *(const bf16x8*)(src + c8 * 8);
            #pragma unroll
            for (int e = 0; e < 8; ++e) Vt[(c8 * 8 + e) * PLD + t] = vv[e];
        }
    }

    // Fragments of q (A) and k (B of q k^T): both contiguous 16B loads
    bf16x16 qf[4], kf[4];
    #pragma unroll
    for (int i = 0; i < 4; ++i) {
        const __bf16* sq = qws + base + (size_t)(i * 16 + r) * DIM_HEAD;
        qf[i] = frag_cat(*(const bf16x8*)(sq + kg), *(const bf16x8*)(sq + 16 + kg));
        const __bf16* sk = kws + base + (size_t)(i * 16 + r) * DIM_HEAD;
        kf[i] = frag_cat(*(const bf16x8*)(sk + kg), *(const bf16x8*)(sk + 16 + kg));
    }

    f32x8 s[4][4] = {};
    #pragma unroll
    for (int i = 0; i < 4; ++i)
        #pragma unroll
        for (int j = 0; j < 4; ++j)
            s[i][j] = wmma_bf16(qf[i], kf[j], s[i][j]);

    // Bias add
#if USE_TDM
    tdm_wait0();
    __syncthreads();   // single-wave: lowers to NOP, orders LDS at IR level
#endif
    #pragma unroll
    for (int i = 0; i < 4; ++i)
        #pragma unroll
        for (int j = 0; j < 4; ++j) {
            int col = j * 16 + r;
            #pragma unroll
            for (int v = 0; v < 8; ++v) {
                int row = i * 16 + 8 * hih + v;
#if USE_TDM
                s[i][j][v] += Bls[row * NTOK + col];
#else
                s[i][j][v] += bp[row * NTOK + col];
#endif
            }
        }

    // Row softmax: each row lives in one 16-lane half at fixed (i,v)
    #pragma unroll
    for (int i = 0; i < 4; ++i) {
        #pragma unroll
        for (int v = 0; v < 8; ++v) {
            float mx = s[i][0][v];
            mx = fmaxf(mx, s[i][1][v]);
            mx = fmaxf(mx, s[i][2][v]);
            mx = fmaxf(mx, s[i][3][v]);
            #pragma unroll
            for (int msk = 1; msk <= 8; msk <<= 1)
                mx = fmaxf(mx, __shfl_xor(mx, msk, 32));
            float sum = 0.f;
            #pragma unroll
            for (int j = 0; j < 4; ++j) {
                float e = __expf(s[i][j][v] - mx);
                s[i][j][v] = e;
                sum += e;
            }
            #pragma unroll
            for (int msk = 1; msk <= 8; msk <<= 1)
                sum += __shfl_xor(sum, msk, 32);
            float inv = 1.0f / sum;
            int row = i * 16 + 8 * hih + v;
            #pragma unroll
            for (int j = 0; j < 4; ++j)
                Ps[row * PLD + j * 16 + r] = (__bf16)(s[i][j][v] * inv);
        }
    }
    __syncthreads();

    // O = P (64x64) * V (64x32)
    f32x8 o[4][2] = {};
    #pragma unroll
    for (int kt = 0; kt < 2; ++kt) {
        bf16x16 pf[4], vf[2];
        #pragma unroll
        for (int i = 0; i < 4; ++i) {
            const __bf16* sp = &Ps[(i * 16 + r) * PLD + kt * 32];
            pf[i] = frag_cat(*(const bf16x8*)(sp + kg), *(const bf16x8*)(sp + 16 + kg));
        }
        #pragma unroll
        for (int j = 0; j < 2; ++j) {
            const __bf16* sv = &Vt[(j * 16 + r) * PLD + kt * 32];
            vf[j] = frag_cat(*(const bf16x8*)(sv + kg), *(const bf16x8*)(sv + 16 + kg));
        }
        #pragma unroll
        for (int i = 0; i < 4; ++i)
            #pragma unroll
            for (int j = 0; j < 2; ++j)
                o[i][j] = wmma_bf16(pf[i], vf[j], o[i][j]);
    }

    // Write attn output to [bwin][tok][h*32+d] (bf16) for the proj GEMM
    #pragma unroll
    for (int i = 0; i < 4; ++i)
        #pragma unroll
        for (int j = 0; j < 2; ++j) {
            int d = h * 32 + j * 16 + r;
            #pragma unroll
            for (int v = 0; v < 8; ++v) {
                int tok = i * 16 + 8 * hih + v;
                aout[((size_t)b * NTOK + tok) * DIM + d] = (__bf16)o[i][j][v];
            }
        }
}

// ---------------------------------------------------------------------------
// Projection GEMM: (Bwin*64, 512) x (512, 512) bf16 -> fp32 out + bias.
// A tile staged by the Tensor Data Mover (with LDS padding); B tile staged
// transposed by the waves. Grid: x = N-chunks (fast), y = M-tiles.
// ---------------------------------------------------------------------------
__global__ __launch_bounds__(256)
void proj_gemm_kernel(const __bf16* __restrict__ aout,
                      const __bf16* __restrict__ wproj_bf,
                      const float* __restrict__ bproj,
                      float* __restrict__ out,
                      unsigned long long rows_total) {
    __shared__ __bf16 As[BM * LDA];
    __shared__ __bf16 Bs[BN * LDA];

    const int tid  = threadIdx.x;
    const int lane = tid & 31;
    const int wid  = tid >> 5;
    const int m0   = blockIdx.y * BM;
    const int n0   = blockIdx.x * BN;
    const int wm   = (wid & 3) * 32;
    const int wn   = (wid >> 2) * 64;
    const int r    = lane & 15;
    const int kg   = (lane >> 4) * 8;
    const int hih  = lane >> 4;

#if USE_TDM
    const unsigned lds_a = (unsigned)(unsigned long long)&As[0];
#endif

    f32x8 acc[2][4] = {};

    for (int k0 = 0; k0 < DIM; k0 += BK) {
#if USE_TDM
        // A tile via TDM: 128 rows x 64B payload + 16B LDS pad per row
        if (wid == 0) {
            tdm_load_2d(lds_a, aout + (size_t)m0 * DIM + k0,
                        /*ds=*/1, /*tile_d0=*/BK, /*tile_d1=*/BM,
                        /*tensor_d0=*/DIM, /*tensor_d1=*/rows_total,
                        /*stride0=*/DIM, /*pad_en=*/1, /*interval=*/3, /*amount=*/3);
        }
#else
        #pragma unroll
        for (int it = 0; it < 2; ++it) {
            int c   = tid + it * 256;      // 0..511
            int row = c >> 2;
            int kc  = (c & 3) * 8;
            bf16x8 v = *(const bf16x8*)(aout + (size_t)(m0 + row) * DIM + k0 + kc);
            *(bf16x8*)&As[row * LDA + kc] = v;
        }
#endif
        // B tile transposed
        #pragma unroll
        for (int it = 0; it < 4; ++it) {
            int c  = tid + it * 256;
            int k  = c >> 5;
            int nb = (c & 31) * 4;
            bf16x4 bv = *(const bf16x4*)(wproj_bf + (size_t)(k0 + k) * DIM + n0 + nb);
            Bs[(nb + 0) * LDA + k] = bv[0];
            Bs[(nb + 1) * LDA + k] = bv[1];
            Bs[(nb + 2) * LDA + k] = bv[2];
            Bs[(nb + 3) * LDA + k] = bv[3];
        }
#if USE_TDM
        if (wid == 0) tdm_wait0();   // issuing wave drains TENSORcnt pre-barrier
#endif
        __syncthreads();

        bf16x16 af[2], bfr[4];
        #pragma unroll
        for (int i = 0; i < 2; ++i) {
            int row = wm + i * 16 + r;
            af[i] = frag_cat(*(const bf16x8*)&As[row * LDA + kg],
                             *(const bf16x8*)&As[row * LDA + 16 + kg]);
        }
        #pragma unroll
        for (int j = 0; j < 4; ++j) {
            int col = wn + j * 16 + r;
            bfr[j] = frag_cat(*(const bf16x8*)&Bs[col * LDA + kg],
                              *(const bf16x8*)&Bs[col * LDA + 16 + kg]);
        }
        #pragma unroll
        for (int i = 0; i < 2; ++i)
            #pragma unroll
            for (int j = 0; j < 4; ++j)
                acc[i][j] = wmma_bf16(af[i], bfr[j], acc[i][j]);
        __syncthreads();
    }

    #pragma unroll
    for (int i = 0; i < 2; ++i)
        #pragma unroll
        for (int j = 0; j < 4; ++j) {
            int col  = n0 + wn + j * 16 + r;
            float bv = bproj[col];
            #pragma unroll
            for (int v = 0; v < 8; ++v) {
                int mrow = m0 + wm + i * 16 + 8 * hih + v;
                out[(size_t)mrow * DIM + col] = acc[i][j][v] + bv;
            }
        }
}

// ---------------------------------------------------------------------------
extern "C" void kernel_launch(void* const* d_in, const int* in_sizes, int n_in,
                              void* d_out, int out_size, void* d_ws, size_t ws_size,
                              hipStream_t stream) {
    (void)n_in; (void)out_size; (void)ws_size;
    const float* x     = (const float*)d_in[0];
    const float* wqkv  = (const float*)d_in[1];
    const float* bqkv  = (const float*)d_in[2];
    const float* table = (const float*)d_in[3];
    const float* wproj = (const float*)d_in[4];
    const float* bproj = (const float*)d_in[5];
    float* out = (float*)d_out;

    const size_t Bwin = (size_t)in_sizes[0] / ((size_t)NTOK * DIM);  // 2048

    char* ws = (char*)d_ws;
    size_t off = 0;
    __bf16* wqkv_bf  = (__bf16*)(ws + off); off += (size_t)DIM * QKVN * 2;
    __bf16* wproj_bf = (__bf16*)(ws + off); off += (size_t)DIM * DIM * 2;
    float*  bias     = (float*)(ws + off);  off += (size_t)HEADS * NTOK * NTOK * 4;
    size_t qkv_elems = Bwin * HEADS * NTOK * DIM_HEAD;
    __bf16* qws  = (__bf16*)(ws + off); off += qkv_elems * 2;
    __bf16* kws  = (__bf16*)(ws + off); off += qkv_elems * 2;
    __bf16* vws  = (__bf16*)(ws + off); off += qkv_elems * 2;
    __bf16* aows = (__bf16*)(ws + off); off += Bwin * NTOK * DIM * 2;

    const int prep_total = DIM * QKVN + DIM * DIM + HEADS * NTOK * NTOK;
    prep_kernel<<<(prep_total + 255) / 256, 256, 0, stream>>>(
        wqkv, wproj, table, wqkv_bf, wproj_bf, bias);

    // N-chunks on x (fast) so blocks sharing an A tile run back-to-back -> L2 reuse
    dim3 g1(QKVN / BN, (unsigned)(Bwin * NTOK / BM));
    qkv_gemm_kernel<<<g1, 256, 0, stream>>>(x, wqkv_bf, bqkv, qws, kws, vws);

    attn_kernel<<<(unsigned)(Bwin * HEADS), 32, 0, stream>>>(qws, kws, vws, bias, aows);

    dim3 g2(DIM / BN, (unsigned)(Bwin * NTOK / BM));
    proj_gemm_kernel<<<g2, 256, 0, stream>>>(aows, wproj_bf, bproj, out,
                                             (unsigned long long)(Bwin * NTOK));
}
